// MultiHeadAttention_54460185313292
// MI455X (gfx1250) — matrix operand order repaired and checked by reading, not compiled, not run on hardware
//
#include <hip/hip_runtime.h>

typedef _Float16 half_t;
typedef __attribute__((ext_vector_type(16))) _Float16 v16h;
typedef __attribute__((ext_vector_type(8)))  _Float16 h8;   // 16 bytes
typedef __attribute__((ext_vector_type(8)))  float    v8f;
typedef __attribute__((ext_vector_type(4)))  float    f4;
typedef __attribute__((ext_vector_type(4)))  unsigned int u32x4;
typedef __attribute__((ext_vector_type(8)))  int      i32x8;
typedef __attribute__((ext_vector_type(4)))  int      i32x4;

union V16 { v16h v; h8 h[2]; };

constexpr int Bb = 4, Tt = 2048, Cc = 1024, Hh = 16, Dd = 64;
constexpr int Mrows = Bb * Tt;           // 8192
constexpr float NEG_BIG = -3.0e38f;

#if defined(__has_builtin)
#if __has_builtin(__builtin_amdgcn_tensor_load_to_lds) && \
    __has_builtin(__builtin_amdgcn_s_wait_tensorcnt)
#define HAVE_TDM 1
#endif
#endif

// ---------------------------------------------------------------------------
// Weight convert + transpose: WT[n*C + k] = (f16) W[k*C + n]
// ---------------------------------------------------------------------------
__global__ __launch_bounds__(256) void k_wT(const float* __restrict__ W,
                                            half_t* __restrict__ WT) {
  int i = blockIdx.x * 256 + threadIdx.x;      // 0 .. C*C-1 exactly
  int n = i >> 10;
  int k = i & (Cc - 1);
  WT[i] = (half_t)W[k * Cc + n];
}

// ---------------------------------------------------------------------------
// QKV projection GEMM: out(f16,[B,H,T,D]) = x(f32,[M,C]) @ WT^T + bias
// Block: 256 threads = 8 waves; tile 256 rows x 64 cols; wave -> 32 rows.
// Double-buffered LDS staging of the weight block.
// ---------------------------------------------------------------------------
__global__ __launch_bounds__(256) void k_qkv(const float* __restrict__ x,
                                             const half_t* __restrict__ WT,
                                             const float* __restrict__ bias,
                                             half_t* __restrict__ out) {
  __shared__ __align__(16) half_t WB[2][64][40];   // 64 cols x 32 k, pitch 40
  const int tid   = threadIdx.x;
  const int wave  = tid >> 5;
  const int lane  = tid & 31;
  const int lhalf = lane >> 4;
  const int l16   = lane & 15;
  const int m0    = blockIdx.x * 256 + wave * 32;
  const int n0    = blockIdx.y * 64;
  const float* xr0 = x + (size_t)(m0 + l16) * Cc;
  const float* xr1 = x + (size_t)(m0 + 16 + l16) * Cc;

  const int scol = tid >> 2;
  const int sseg = (tid & 3) * 8;
  const half_t* wsrc = WT + (size_t)(n0 + scol) * Cc + sseg;

  // prologue: stage k-block 0 into buffer 0
  *(h8*)&WB[0][scol][sseg] = *(const h8*)(wsrc);
  __syncthreads();

  v8f acc[2][4] = {};
  for (int k0 = 0; k0 < Cc; k0 += 32) {
    const int p = (k0 >> 5) & 1;
    if (k0 + 32 < Cc) {                      // stage next block into other buf
      *(h8*)&WB[p ^ 1][scol][sseg] = *(const h8*)(wsrc + k0 + 32);
      __builtin_prefetch(xr0 + k0 + 32, 0, 1);
      __builtin_prefetch(xr1 + k0 + 32, 0, 1);
    }

    // A fragments (f32 -> f16): e0..7 -> k=k0+lhalf*8+e ; e8..15 -> +16
    V16 a[2];
#pragma unroll
    for (int ms = 0; ms < 2; ++ms) {
      const float* ap = (ms ? xr1 : xr0) + k0 + lhalf * 8;
      f4 a0 = *(const f4*)(ap);
      f4 a1 = *(const f4*)(ap + 4);
      f4 a2 = *(const f4*)(ap + 16);
      f4 a3 = *(const f4*)(ap + 20);
#pragma unroll
      for (int e = 0; e < 4; ++e) {
        a[ms].v[e]      = (half_t)a0[e];
        a[ms].v[4 + e]  = (half_t)a1[e];
        a[ms].v[8 + e]  = (half_t)a2[e];
        a[ms].v[12 + e] = (half_t)a3[e];
      }
    }

#pragma unroll
    for (int ct = 0; ct < 4; ++ct) {
      V16 b;
      const half_t* bp = &WB[p][ct * 16 + l16][lhalf * 8];
      b.h[0] = *(const h8*)bp;
      b.h[1] = *(const h8*)(bp + 16);
      acc[0][ct] = __builtin_amdgcn_wmma_f32_16x16x32_f16(
          false, a[0].v, false, b.v, (short)0, acc[0][ct], false, false);
      acc[1][ct] = __builtin_amdgcn_wmma_f32_16x16x32_f16(
          false, a[1].v, false, b.v, (short)0, acc[1][ct], false, false);
    }
    __syncthreads();
  }

  // epilogue: scatter to [B,H,T,D] f16 with bias
#pragma unroll
  for (int ms = 0; ms < 2; ++ms) {
#pragma unroll
    for (int ct = 0; ct < 4; ++ct) {
      int c  = n0 + ct * 16 + l16;
      float bv = bias[c];
      int hh = c >> 6;
      int dd = c & 63;
#pragma unroll
      for (int vj = 0; vj < 8; ++vj) {
        int m  = m0 + ms * 16 + vj + 8 * lhalf;
        int bb = m >> 11;
        int tt = m & (Tt - 1);
        out[(((size_t)(bb * Hh + hh) * Tt + tt) << 6) + dd] =
            (half_t)(acc[ms][ct][vj] + bv);
      }
    }
  }
}

// ---------------------------------------------------------------------------
// Flash attention (causal). Block: 128 threads = 4 waves, owns 64 q-rows of
// one (b,h). K tile staged via Tensor Data Mover (with LDS row padding),
// V tile staged transposed manually. Wave w -> q rows q0+16w..+15.
// ---------------------------------------------------------------------------
__global__ __launch_bounds__(128) void k_attn(const half_t* __restrict__ Q,
                                              const half_t* __restrict__ K,
                                              const half_t* __restrict__ V,
                                              half_t* __restrict__ Y) {
  __shared__ __align__(16) half_t Kt[64][72];       // [key][d], pad 8 halves
  __shared__ __align__(16) half_t Vt[64][72];       // [d][key], padded
  __shared__ __align__(16) half_t Pb[4][16][40];    // per-wave P, padded

  const int tid   = threadIdx.x;
  const int wave  = tid >> 5;
  const int lane  = tid & 31;
  const int lhalf = lane >> 4;
  const int l16   = lane & 15;
  const int bh    = blockIdx.x;            // b*H + h
  const int b     = bh >> 4;
  const int h     = bh & 15;
  const int q0    = blockIdx.y * 64;
  const size_t headBase = (size_t)bh * Tt * Dd;

  // Q fragments (2 K-steps of 32 over D=64), loaded once
  V16 qf[2];
  {
    const half_t* qp = Q + headBase + (size_t)(q0 + wave * 16 + l16) * Dd;
#pragma unroll
    for (int s = 0; s < 2; ++s) {
      qf[s].h[0] = *(const h8*)(qp + s * 32 + lhalf * 8);
      qf[s].h[1] = *(const h8*)(qp + s * 32 + 16 + lhalf * 8);
    }
  }

  float mrun[8], lrun[8];
  v8f o[4] = {};
#pragma unroll
  for (int vj = 0; vj < 8; ++vj) { mrun[vj] = NEG_BIG; lrun[vj] = 0.f; }

  for (int j0 = 0; j0 <= q0; j0 += 64) {
    __syncthreads();  // prior tile fully consumed before overwrite

#ifdef HAVE_TDM
    // ---- K tile via Tensor Data Mover: 64x64 2-byte tile, row pitch 64,
    //      LDS padding 4 DWORDs every 32 DWORDs (matches Kt[64][72]).
    if (wave == 0) {
      unsigned ldsKt = (unsigned)(unsigned long long)(void*)&Kt[0][0];
      unsigned long long ga =
          (unsigned long long)(K + headBase + (size_t)j0 * Dd);
      u32x4 g0;
      g0[0] = 1u;                                   // count=1, user mode
      g0[1] = ldsKt;                                // lds_addr
      g0[2] = (unsigned)(ga & 0xffffffffu);         // global_addr[31:0]
      g0[3] = (unsigned)((ga >> 32) & 0x1ffffffu) | (2u << 30); // type=2
      i32x8 g1;
      g1[0] = (1 << 16) |                           // data_size = 2 bytes
              (1 << 20) |                           // pad_enable
              (4 << 22) |                           // pad_interval: 32 DWORDs
              (3 << 25);                            // pad_amount: 4 DWORDs
      g1[1] = (64 << 16);                           // tensor_dim0 = 64
      g1[2] = (64 << 16);                           // tensor_dim1 = 64
      g1[3] = (64 << 16);                           // tile_dim0 = 64
      g1[4] = 64;                                   // tile_dim1 = 64
      g1[5] = 64;                                   // tensor_dim0_stride = 64
      g1[6] = 0;
      g1[7] = 0;
      i32x4 gz4 = {0, 0, 0, 0};
      i32x8 gz8 = {0, 0, 0, 0, 0, 0, 0, 0};
      __builtin_amdgcn_tensor_load_to_lds(g0, g1, gz4, gz4, gz8, 0);
    }
#endif

    // ---- V tile: cooperative transposed staging (TDM cannot transpose)
    {
      int r  = tid >> 1;
      int cb = (tid & 1) * 32;
      const half_t* vp = V + headBase + (size_t)(j0 + r) * Dd + cb;
#pragma unroll
      for (int i = 0; i < 4; ++i) {
        h8 t = *(const h8*)(vp + i * 8);
#pragma unroll
        for (int e = 0; e < 8; ++e) Vt[cb + i * 8 + e][r] = t[e];
      }
#ifndef HAVE_TDM
      const half_t* kp = K + headBase + (size_t)(j0 + r) * Dd + cb;
#pragma unroll
      for (int i = 0; i < 4; ++i)
        *(h8*)&Kt[r][cb + i * 8] = *(const h8*)(kp + i * 8);
#endif
    }
#ifdef HAVE_TDM
    if (wave == 0) __builtin_amdgcn_s_wait_tensorcnt(0);
#endif
    __syncthreads();

    // two 32-key blocks per tile
#pragma unroll
    for (int kb = 0; kb < 2; ++kb) {
      const int jbase = kb * 32;
      v8f S[2];
#pragma unroll
      for (int sub = 0; sub < 2; ++sub) {
        const int jloc = jbase + sub * 16;
        V16 bf0, bf1;               // preload both K fragments, then 2 WMMA
        {
          const half_t* bp0 = &Kt[jloc + l16][lhalf * 8];
          const half_t* bp1 = &Kt[jloc + l16][32 + lhalf * 8];
          bf0.h[0] = *(const h8*)bp0;
          bf0.h[1] = *(const h8*)(bp0 + 16);
          bf1.h[0] = *(const h8*)bp1;
          bf1.h[1] = *(const h8*)(bp1 + 16);
        }
        v8f s = {};
        s = __builtin_amdgcn_wmma_f32_16x16x32_f16(
            false, qf[0].v, false, bf0.v, (short)0, s, false, false);
        s = __builtin_amdgcn_wmma_f32_16x16x32_f16(
            false, qf[1].v, false, bf1.v, (short)0, s, false, false);
        // scale + causal mask
#pragma unroll
        for (int vj = 0; vj < 8; ++vj) {
          int mg = q0 + wave * 16 + vj + 8 * lhalf;
          int jg = j0 + jloc + l16;
          float val = s[vj] * 0.125f;           // 1/sqrt(64)
          s[vj] = (jg <= mg) ? val : NEG_BIG;
        }
        S[sub] = s;
      }

      // online softmax update over this 32-key block
#pragma unroll
      for (int vj = 0; vj < 8; ++vj) {
        float bm = fmaxf(S[0][vj], S[1][vj]);
#pragma unroll
        for (int off = 8; off; off >>= 1)
          bm = fmaxf(bm, __shfl_xor(bm, off, 32));
        float nm    = fmaxf(mrun[vj], bm);
        float alpha = __expf(mrun[vj] - nm);
        mrun[vj]    = nm;
        float p0 = __expf(S[0][vj] - nm);
        float p1 = __expf(S[1][vj] - nm);
        float rs = p0 + p1;
#pragma unroll
        for (int off = 8; off; off >>= 1)
          rs += __shfl_xor(rs, off, 32);
        lrun[vj] = lrun[vj] * alpha + rs;
#pragma unroll
        for (int ct = 0; ct < 4; ++ct) o[ct][vj] *= alpha;
        int m = vj + 8 * lhalf;
        Pb[wave][m][l16]      = (half_t)p0;
        Pb[wave][m][16 + l16] = (half_t)p1;
      }

      // P (16x32, per-wave LDS) @ V (32x64): preload 4 fragments, then 4 WMMA
      V16 pa;
      {
        const half_t* pr = &Pb[wave][l16][0];
        pa.h[0] = *(const h8*)(pr + lhalf * 8);
        pa.h[1] = *(const h8*)(pr + 16 + lhalf * 8);
      }
      V16 vb[4];
#pragma unroll
      for (int ct = 0; ct < 4; ++ct) {
        const half_t* vbp = &Vt[ct * 16 + l16][jbase + lhalf * 8];
        vb[ct].h[0] = *(const h8*)vbp;
        vb[ct].h[1] = *(const h8*)(vbp + 16);
      }
#pragma unroll
      for (int ct = 0; ct < 4; ++ct)
        o[ct] = __builtin_amdgcn_wmma_f32_16x16x32_f16(
            false, pa.v, false, vb[ct].v, (short)0, o[ct], false, false);
    }
  }

  // epilogue: O / l -> Y (f16, [B*T, C] with col = h*64 + d)
#pragma unroll
  for (int ct = 0; ct < 4; ++ct) {
#pragma unroll
    for (int vj = 0; vj < 8; ++vj) {
      int mg = q0 + wave * 16 + vj + 8 * lhalf;
      int c  = h * 64 + ct * 16 + l16;
      Y[((size_t)(b * Tt + mg)) * Cc + c] = (half_t)(o[ct][vj] / lrun[vj]);
    }
  }
}

// ---------------------------------------------------------------------------
// Output projection: d_out(f32,[M,C]) = Y(f16) @ WpT^T + bp
// Same 256x64 tiling / double-buffered staging as k_qkv.
// ---------------------------------------------------------------------------
__global__ __launch_bounds__(256) void k_proj(const half_t* __restrict__ Yh,
                                              const half_t* __restrict__ WT,
                                              const float* __restrict__ bias,
                                              float* __restrict__ out) {
  __shared__ __align__(16) half_t WB[2][64][40];
  const int tid   = threadIdx.x;
  const int wave  = tid >> 5;
  const int lane  = tid & 31;
  const int lhalf = lane >> 4;
  const int l16   = lane & 15;
  const int m0    = blockIdx.x * 256 + wave * 32;
  const int n0    = blockIdx.y * 64;
  const half_t* ar0 = Yh + (size_t)(m0 + l16) * Cc;
  const half_t* ar1 = Yh + (size_t)(m0 + 16 + l16) * Cc;

  const int scol = tid >> 2;
  const int sseg = (tid & 3) * 8;
  const half_t* wsrc = WT + (size_t)(n0 + scol) * Cc + sseg;

  *(h8*)&WB[0][scol][sseg] = *(const h8*)(wsrc);
  __syncthreads();

  v8f acc[2][4] = {};
  for (int k0 = 0; k0 < Cc; k0 += 32) {
    const int p = (k0 >> 5) & 1;
    if (k0 + 32 < Cc) {
      *(h8*)&WB[p ^ 1][scol][sseg] = *(const h8*)(wsrc + k0 + 32);
      __builtin_prefetch(ar0 + k0 + 32, 0, 1);
      __builtin_prefetch(ar1 + k0 + 32, 0, 1);
    }

    V16 a[2];
    a[0].h[0] = *(const h8*)(ar0 + k0 + lhalf * 8);
    a[0].h[1] = *(const h8*)(ar0 + k0 + 16 + lhalf * 8);
    a[1].h[0] = *(const h8*)(ar1 + k0 + lhalf * 8);
    a[1].h[1] = *(const h8*)(ar1 + k0 + 16 + lhalf * 8);

#pragma unroll
    for (int ct = 0; ct < 4; ++ct) {
      V16 b;
      const half_t* bp = &WB[p][ct * 16 + l16][lhalf * 8];
      b.h[0] = *(const h8*)bp;
      b.h[1] = *(const h8*)(bp + 16);
      acc[0][ct] = __builtin_amdgcn_wmma_f32_16x16x32_f16(
          false, a[0].v, false, b.v, (short)0, acc[0][ct], false, false);
      acc[1][ct] = __builtin_amdgcn_wmma_f32_16x16x32_f16(
          false, a[1].v, false, b.v, (short)0, acc[1][ct], false, false);
    }
    __syncthreads();
  }

#pragma unroll
  for (int ms = 0; ms < 2; ++ms) {
#pragma unroll
    for (int ct = 0; ct < 4; ++ct) {
      int c = n0 + ct * 16 + l16;
      float bv = bias[c];
#pragma unroll
      for (int vj = 0; vj < 8; ++vj) {
        int m = m0 + ms * 16 + vj + 8 * lhalf;
        out[(size_t)m * Cc + c] = acc[ms][ct][vj] + bv;
      }
    }
  }
}

// ---------------------------------------------------------------------------
extern "C" void kernel_launch(void* const* d_in, const int* in_sizes, int n_in,
                              void* d_out, int out_size, void* d_ws, size_t ws_size,
                              hipStream_t stream) {
  (void)in_sizes; (void)n_in; (void)out_size; (void)ws_size;
  const float* x  = (const float*)d_in[0];
  const float* Wq = (const float*)d_in[1];
  const float* bq = (const float*)d_in[2];
  const float* Wk = (const float*)d_in[3];
  const float* bk = (const float*)d_in[4];
  const float* Wv = (const float*)d_in[5];
  const float* bv = (const float*)d_in[6];
  const float* Wp = (const float*)d_in[7];
  const float* bp = (const float*)d_in[8];

  char* ws = (char*)d_ws;
  const size_t WBYTES = (size_t)Cc * Cc * sizeof(half_t);      // 2 MiB
  const size_t TBYTES = (size_t)Mrows * Cc * sizeof(half_t);   // 16 MiB
  half_t* WqT = (half_t*)(ws + 0 * WBYTES);
  half_t* WkT = (half_t*)(ws + 1 * WBYTES);
  half_t* WvT = (half_t*)(ws + 2 * WBYTES);
  half_t* WpT = (half_t*)(ws + 3 * WBYTES);
  half_t* Qh  = (half_t*)(ws + 4 * WBYTES);
  half_t* Kh  = (half_t*)(ws + 4 * WBYTES + 1 * TBYTES);
  half_t* Vh  = (half_t*)(ws + 4 * WBYTES + 2 * TBYTES);
  half_t* Yh  = (half_t*)(ws + 4 * WBYTES + 3 * TBYTES);

  dim3 cg(Cc * Cc / 256);
  k_wT<<<cg, 256, 0, stream>>>(Wq, WqT);
  k_wT<<<cg, 256, 0, stream>>>(Wk, WkT);
  k_wT<<<cg, 256, 0, stream>>>(Wv, WvT);
  k_wT<<<cg, 256, 0, stream>>>(Wp, WpT);

  dim3 gg(Mrows / 256, Cc / 64);
  k_qkv<<<gg, 256, 0, stream>>>(x, WqT, bq, Qh);
  k_qkv<<<gg, 256, 0, stream>>>(x, WkT, bk, Kh);
  k_qkv<<<gg, 256, 0, stream>>>(x, WvT, bv, Vh);

  dim3 ga(Bb * Hh, Tt / 64);
  k_attn<<<ga, 128, 0, stream>>>(Qh, Kh, Vh, Yh);

  k_proj<<<gg, 256, 0, stream>>>(Yh, WpT, bp, (float*)d_out);
}
